// SPINN_44942537786058
// MI455X (gfx1250) — matrix operand order repaired and checked by reading, not compiled, not run on hardware
//
#include <hip/hip_runtime.h>
#include <hip/hip_bf16.h>
#include <math.h>

typedef __attribute__((ext_vector_type(16))) __bf16 v16bf;
typedef __attribute__((ext_vector_type(8)))  float  v8f;
typedef __attribute__((ext_vector_type(4)))  unsigned int v4u;
typedef __attribute__((ext_vector_type(8)))  int v8i;
typedef __attribute__((ext_vector_type(4)))  int v4i;

#define L_SENT   128
#define BATCH    256
#define DMODEL   256
#define TTRACK   128
#define NSTEPS   255   // 2L-1
#define ACT_SHIFT  1
#define ACT_REDUCE 2
#define CPOOL    16    // composed-node ring (live composed depth << 16 for canonical S/R seq)

// ---------------- device helpers ----------------
__device__ __forceinline__ float sigf(float x) { return 1.f / (1.f + __expf(-x)); }

// stack entry value: idx<0 -> stack_base row; idx<L -> buffers[idx]; else composed ring
__device__ __forceinline__ float stackval(int idx, int b, int d,
                                          const float* buffers, const float* stack_base,
                                          const float* composed) {
  if (idx < 0)      return stack_base[d];
  if (idx < L_SENT) return buffers[((size_t)idx * BATCH + b) * (2 * DMODEL) + d];
  return composed[(((size_t)((idx - L_SENT) & (CPOOL - 1))) * BATCH + b) * (2 * DMODEL) + d];
}

// Issue one TDM 1-D copy: `words` 4-byte words from gaddr -> LDS ldsAddr.
// D# per CDNA5 ISA ch8: group0 {count=1, lds_addr, global_addr[56:0], type=2},
// group1 {data_size=4B, tensor_dim0=tile_dim0=words, tensor_dim1=1, dim0_stride=words},
// groups 2/3 zero (<=2D tensor). This toolchain exposes the 6-arg builtin
// (g0, g1, g2, g3, extra group, cpol).
__device__ __forceinline__ void tdm_copy_1d(unsigned ldsAddr, unsigned long long gaddr,
                                            unsigned words) {
  v4u g0 = { 1u, ldsAddr, (unsigned)gaddr,
             (unsigned)((gaddr >> 32) & 0x01FFFFFFu) | 0x80000000u };
  v8i g1 = { (int)(2u << 16),                                   // data_size=4B, mask=0
             (int)((words & 0xFFFFu) << 16),                    // tensor_dim0[15:0]
             (int)(((words >> 16) & 0xFFFFu) | (1u << 16)),     // tensor_dim0[31:16], tensor_dim1=1
             (int)(words << 16),                                // tile_dim0 (words <= 65535)
             0,                                                 // tile_dim1/2 unused
             (int)words,                                        // tensor_dim0_stride[31:0]
             0, 0 };
  v4i z4 = {0, 0, 0, 0};
  v8i z8 = {0, 0, 0, 0, 0, 0, 0, 0};
  __builtin_amdgcn_tensor_load_to_lds(g0, g1, z4, z4, z8, 0);
}

// ---------------- one-time weight pre-swizzle into WMMA B-fragment layout ----------------
// B 32x16 bf16 fragment (ISA 7.12.2): lane l holds column n=l&15; k = (l<16?0:8) + i + (i>=8?8:0), i=0..15
__global__ void k_swizzle_w(const float* __restrict__ W, int K, int N, int KTtot, int ktOff,
                            __bf16* __restrict__ hi, __bf16* __restrict__ lo) {
  int Kt = K >> 5, Nt = N >> 4;
  int tid = blockIdx.x * blockDim.x + threadIdx.x;
  if (tid >= Kt * Nt * 32) return;
  int lane = tid & 31;
  int rest = tid >> 5;
  int ktl  = rest % Kt;
  int nt   = rest / Kt;
  int n    = nt * 16 + (lane & 15);
  int kbase = ktl * 32 + ((lane < 16) ? 0 : 8);
  size_t fr = ((size_t)(nt * KTtot + ktOff + ktl) * 32 + lane) * 16;
  for (int i = 0; i < 16; ++i) {
    float v  = W[(size_t)(kbase + i + ((i >= 8) ? 8 : 0)) * N + n];
    __bf16 h = (__bf16)v;
    hi[fr + i] = h;
    lo[fr + i] = (__bf16)(v - (float)h);
  }
}

// ---------------- per-step A-fragment swizzles ----------------
// A 16x32 bf16 fragment (ISA 7.12.2): lane l holds row m=l&15; k = (l<16?0:8) + i + (i>=8?8:0)
__global__ void k_swizA_tracker(const float* __restrict__ buffers, const float* __restrict__ sentence_end,
                                const float* __restrict__ stack_base, const float* __restrict__ composed,
                                const float* __restrict__ h, const int* __restrict__ state,
                                __bf16* __restrict__ Ah, __bf16* __restrict__ Al) {
  const int KT = 28; // K = 256(buf_h)+256(s1_h)+256(s2_h)+128(h) = 896
  int tid = blockIdx.x * blockDim.x + threadIdx.x;
  if (tid >= 16 * KT * 32) return;
  int lane = tid & 31;
  int rest = tid >> 5;
  int kt = rest % KT;
  int mt = rest / KT;
  int m  = mt * 16 + (lane & 15);
  int kb = (lane < 16) ? 0 : 8;
  int sp = state[0], ptr = state[1];
  int s1 = state[3 + sp - 1];
  int s2 = state[3 + sp - 2];
  size_t fr = ((size_t)(mt * KT + kt) * 32 + lane) * 16;
  for (int i = 0; i < 16; ++i) {
    int kk = kt * 32 + kb + i + ((i >= 8) ? 8 : 0);
    float v;
    if (kk < 256)      v = (ptr < L_SENT) ? buffers[((size_t)ptr * BATCH + m) * 512 + kk]
                                          : sentence_end[kk];
    else if (kk < 512) v = stackval(s1, m, kk - 256, buffers, stack_base, composed);
    else if (kk < 768) v = stackval(s2, m, kk - 512, buffers, stack_base, composed);
    else               v = h[m * TTRACK + (kk - 768)];
    __bf16 hv = (__bf16)v;
    Ah[fr + i] = hv;
    Al[fr + i] = (__bf16)(v - (float)hv);
  }
}

__global__ void k_swizA_reduce(const float* __restrict__ buffers, const float* __restrict__ stack_base,
                               const float* __restrict__ composed, const float* __restrict__ h,
                               const int* __restrict__ state, const int* __restrict__ act,
                               __bf16* __restrict__ Ah, __bf16* __restrict__ Al) {
  if (act[0] != ACT_REDUCE) return;
  const int KT = 20; // K = 256(left_h)+256(right_h)+128(h) = 640
  int tid = blockIdx.x * blockDim.x + threadIdx.x;
  if (tid >= 16 * KT * 32) return;
  int lane = tid & 31;
  int rest = tid >> 5;
  int kt = rest % KT;
  int mt = rest / KT;
  int m  = mt * 16 + (lane & 15);
  int kb = (lane < 16) ? 0 : 8;
  int sp = state[0];
  int s1 = state[3 + sp - 1]; // right = stack[-1]
  int s2 = state[3 + sp - 2]; // left  = stack[-2]
  size_t fr = ((size_t)(mt * KT + kt) * 32 + lane) * 16;
  for (int i = 0; i < 16; ++i) {
    int kk = kt * 32 + kb + i + ((i >= 8) ? 8 : 0);
    float v;
    if (kk < 256)      v = stackval(s2, m, kk,        buffers, stack_base, composed);
    else if (kk < 512) v = stackval(s1, m, kk - 256,  buffers, stack_base, composed);
    else               v = h[m * TTRACK + (kk - 512)];
    __bf16 hv = (__bf16)v;
    Ah[fr + i] = hv;
    Al[fr + i] = (__bf16)(v - (float)hv);
  }
}

// ---------------- step GEMM with TDM-staged A fragments in LDS ----------------
// Block = 8 waves sharing one m-tile (A frags staged once via tensor_load_to_lds),
// each wave owns one n-tile. out = A*B + bias0 + bias1, bf16 hi/lo split x3 WMMA.
template <int KT>
__global__ void k_gemm_tdm(const __bf16* __restrict__ Ah, const __bf16* __restrict__ Al,
                           const __bf16* __restrict__ Bh, const __bf16* __restrict__ Bl,
                           int Mt, int Nt,
                           const float* __restrict__ bias0, const float* __restrict__ bias1,
                           float* __restrict__ out, int N,
                           const int* __restrict__ act, int req) {
  __shared__ __bf16 sAh[KT * 512];
  __shared__ __bf16 sAl[KT * 512];
  if (req >= 0 && act[0] != req) return;
  int ngroups = Nt >> 3;
  int mt   = blockIdx.x / ngroups;
  int ngrp = blockIdx.x % ngroups;
  int wave = threadIdx.x >> 5;
  int lane = threadIdx.x & 31;
  int nt   = ngrp * 8 + wave;

  if (threadIdx.x == 0) {
    const unsigned words = KT * 256; // 4-byte words per fragment array for one m-tile
    tdm_copy_1d((unsigned)(size_t)(void*)sAh,
                (unsigned long long)(Ah + (size_t)mt * KT * 512), words);
    tdm_copy_1d((unsigned)(size_t)(void*)sAl,
                (unsigned long long)(Al + (size_t)mt * KT * 512), words);
  }
  __builtin_amdgcn_s_wait_tensorcnt(0); // NOP for non-issuing waves (TENSORcnt==0)
  __syncthreads();

  v8f acc = {};
  for (int kt = 0; kt < KT; ++kt) {
    size_t fb = ((size_t)(nt * KT + kt) * 32 + lane) * 16;
    if (kt + 1 < KT) __builtin_prefetch(Bh + fb + 512, 0, 1);
    v16bf ah = *(const v16bf*)(sAh + (kt * 32 + lane) * 16); // ds_load_b128 x2
    v16bf al = *(const v16bf*)(sAl + (kt * 32 + lane) * 16);
    v16bf bh = *(const v16bf*)(Bh + fb);
    v16bf bl = *(const v16bf*)(Bl + fb);
    acc = __builtin_amdgcn_wmma_f32_16x16x32_bf16(false, ah, false, bh, (short)0, acc, false, false);
    acc = __builtin_amdgcn_wmma_f32_16x16x32_bf16(false, ah, false, bl, (short)0, acc, false, false);
    acc = __builtin_amdgcn_wmma_f32_16x16x32_bf16(false, al, false, bh, (short)0, acc, false, false);
  }
  int n    = nt * 16 + (lane & 15);
  int mrow = mt * 16 + ((lane < 16) ? 0 : 8);
  float bs = (bias0 ? bias0[n] : 0.f) + (bias1 ? bias1[n] : 0.f);
  for (int r = 0; r < 8; ++r)
    out[(size_t)(mrow + r) * N + n] = acc[r] + bs;
}

// ---------------- big projection GEMM: buffers = sentences @ W_proj + b_proj ----------------
// A fragments built inline (fp32 source, hi/lo split in regs); 4 N-tiles per wave for A reuse
__global__ void k_proj(const float* __restrict__ A, const __bf16* __restrict__ Bh,
                       const __bf16* __restrict__ Bl, const float* __restrict__ bias,
                       float* __restrict__ out) {
  const int KT = 16; // K=512
  int wid  = blockIdx.x * (blockDim.x >> 5) + (threadIdx.x >> 5);
  int lane = threadIdx.x & 31;
  int mt = wid >> 3, ng = wid & 7; // 2048 m-tiles x 8 groups of 4 n-tiles
  if (mt >= 2048) return;
  int m  = mt * 16 + (lane & 15);
  int kb = (lane < 16) ? 0 : 8;
  v8f acc[4] = {};
  for (int kt = 0; kt < KT; ++kt) {
    v16bf ah, al;
    for (int i = 0; i < 16; ++i) {
      int kk = kt * 32 + kb + i + ((i >= 8) ? 8 : 0);
      float v   = A[(size_t)m * 512 + kk];
      __bf16 hv = (__bf16)v;
      ah[i] = hv;
      al[i] = (__bf16)(v - (float)hv);
    }
    for (int j = 0; j < 4; ++j) {
      int nt = ng * 4 + j;
      size_t fb = ((size_t)(nt * KT + kt) * 32 + lane) * 16;
      v16bf bh = *(const v16bf*)(Bh + fb);
      v16bf bl = *(const v16bf*)(Bl + fb);
      acc[j] = __builtin_amdgcn_wmma_f32_16x16x32_bf16(false, ah, false, bh, (short)0, acc[j], false, false);
      acc[j] = __builtin_amdgcn_wmma_f32_16x16x32_bf16(false, ah, false, bl, (short)0, acc[j], false, false);
      acc[j] = __builtin_amdgcn_wmma_f32_16x16x32_bf16(false, al, false, bh, (short)0, acc[j], false, false);
    }
  }
  int mrow = mt * 16 + ((lane < 16) ? 0 : 8);
  for (int j = 0; j < 4; ++j) {
    int n   = (ng * 4 + j) * 16 + (lane & 15);
    float b = bias[n];
    for (int r = 0; r < 8; ++r)
      out[(size_t)(mrow + r) * 512 + n] = acc[j][r] + b;
  }
}

// ---------------- LSTMCell gate nonlinearity + h,c update + softmax CE loss ----------------
__global__ void k_update(const float* __restrict__ gates, float* __restrict__ h, float* __restrict__ c,
                         const float* __restrict__ W_trans, const float* __restrict__ b_trans,
                         const int* __restrict__ act, float* __restrict__ loss) {
  __shared__ float red[3 * TTRACK];
  int b = blockIdx.x, j = threadIdx.x;
  const float* g = gates + (size_t)b * 512;
  float ig = g[j], fg = g[128 + j], gg = g[256 + j], og = g[384 + j];
  float cn = sigf(fg) * c[b * TTRACK + j] + sigf(ig) * tanhf(gg);
  float hn = sigf(og) * tanhf(cn);
  c[b * TTRACK + j] = cn;
  h[b * TTRACK + j] = hn;
  red[j]           = hn * W_trans[j * 3 + 0];
  red[128 + j]     = hn * W_trans[j * 3 + 1];
  red[256 + j]     = hn * W_trans[j * 3 + 2];
  __syncthreads();
  if (j == 0) {
    float l0 = b_trans[0], l1 = b_trans[1], l2 = b_trans[2];
    for (int t = 0; t < 128; ++t) { l0 += red[t]; l1 += red[128 + t]; l2 += red[256 + t]; }
    float mx  = fmaxf(l0, fmaxf(l1, l2));
    float lse = mx + __logf(__expf(l0 - mx) + __expf(l1 - mx) + __expf(l2 - mx));
    int a = act[0];
    float la = (a == 0) ? l0 : ((a == 1) ? l1 : l2);
    unsafeAtomicAdd(loss, lse - la);
  }
}

// ---------------- TreeLSTM composition (reduce steps only) ----------------
__global__ void k_compose(const float* __restrict__ lstm_in, const float* __restrict__ buffers,
                          const float* __restrict__ stack_base, float* __restrict__ composed,
                          const int* __restrict__ state, const int* __restrict__ act) {
  if (act[0] != ACT_REDUCE) return;
  int b = blockIdx.x, d = threadIdx.x;
  int sp = state[0];
  int s1 = state[3 + sp - 1]; // right
  int s2 = state[3 + sp - 2]; // left
  int rc = state[2] & (CPOOL - 1);
  const float* li = lstm_in + (size_t)b * 1280;
  float ag = li[d], ig = li[256 + d], f1 = li[512 + d], f2 = li[768 + d], og = li[1024 + d];
  float lc  = stackval(s2, b, 256 + d, buffers, stack_base, composed);
  float rcv = stackval(s1, b, 256 + d, buffers, stack_base, composed);
  float cn = tanhf(ag) * sigf(ig) + sigf(f1) * lc + sigf(f2) * rcv;
  float hn = sigf(og) * tanhf(cn);
  composed[((size_t)rc * BATCH + b) * 512 + d]       = hn;
  composed[((size_t)rc * BATCH + b) * 512 + 256 + d] = cn;
}

// ---------------- stack bookkeeping (1 thread) ----------------
__global__ void k_book(int* __restrict__ state, const int* __restrict__ act) {
  int a  = act[0];
  int sp = state[0];
  if (a == ACT_SHIFT) {
    state[3 + sp] = state[1];  // push buffer index
    state[0] = sp + 1;
    state[1] = state[1] + 1;   // ptr++
  } else {
    int rc = state[2];
    state[3 + sp - 2] = L_SENT + (rc & (CPOOL - 1)); // pop,pop,push composed
    state[0] = sp - 1;
    state[2] = rc + 1;
  }
}

// ---------------- init / final ----------------
__global__ void k_init(float* __restrict__ h, float* __restrict__ c,
                       const float* __restrict__ context_begin,
                       int* __restrict__ state, float* __restrict__ loss) {
  int b = blockIdx.x, j = threadIdx.x;
  h[b * TTRACK + j] = context_begin[j];
  c[b * TTRACK + j] = context_begin[TTRACK + j];
  if (b == 0 && j == 0) {
    state[0] = 2;  // sp
    state[1] = 0;  // ptr
    state[2] = 0;  // rcount
    state[3] = -1; // stack_base
    state[4] = -1; // stack_base
    loss[0]  = 0.f;
  }
}

__global__ void k_final(const int* __restrict__ state, const float* __restrict__ buffers,
                        const float* __restrict__ stack_base, const float* __restrict__ composed,
                        const float* __restrict__ loss, float* __restrict__ out) {
  int b = blockIdx.x, d = threadIdx.x;
  int sp  = state[0];
  int top = state[3 + sp - 1];
  out[(size_t)b * DMODEL + d] = stackval(top, b, d, buffers, stack_base, composed);
  if (b == 0 && d == 0) out[65536] = loss[0] / (float)(NSTEPS * BATCH);
}

// ---------------- host ----------------
extern "C" void kernel_launch(void* const* d_in, const int* in_sizes, int n_in,
                              void* d_out, int out_size, void* d_ws, size_t ws_size,
                              hipStream_t stream) {
  const float* sentences     = (const float*)d_in[0];
  const float* W_proj        = (const float*)d_in[1];
  const float* b_proj        = (const float*)d_in[2];
  const float* W_left        = (const float*)d_in[3];
  const float* b_left        = (const float*)d_in[4];
  const float* W_right       = (const float*)d_in[5];
  const float* W_track       = (const float*)d_in[6];
  const float* W_ih          = (const float*)d_in[7];
  const float* b_ih          = (const float*)d_in[8];
  const float* W_hh          = (const float*)d_in[9];
  const float* b_hh          = (const float*)d_in[10];
  const float* W_trans       = (const float*)d_in[11];
  const float* b_trans       = (const float*)d_in[12];
  const float* sentence_end  = (const float*)d_in[13];
  const float* stack_base    = (const float*)d_in[14];
  const float* context_begin = (const float*)d_in[15];
  const int*   gold          = (const int*)d_in[16];

  // workspace carving
  char* p = (char*)d_ws;
  auto alloc = [&](size_t bytes) -> void* {
    void* r = (void*)p;
    p += (bytes + 255) & ~(size_t)255;
    return r;
  };
  float* buffers  = (float*)alloc((size_t)L_SENT * BATCH * 512 * 4); // 64 MB
  float* composed = (float*)alloc((size_t)CPOOL * BATCH * 512 * 4);  // 8 MB ring
  float* gates    = (float*)alloc((size_t)BATCH * 512 * 4);
  float* lstm_in  = (float*)alloc((size_t)BATCH * 1280 * 4);
  float* hbuf     = (float*)alloc((size_t)BATCH * TTRACK * 4);
  float* cbuf     = (float*)alloc((size_t)BATCH * TTRACK * 4);
  float* loss     = (float*)alloc(256);
  int*   state    = (int*)alloc(1024);
  // weight fragments (bf16 hi/lo), one-time swizzle
  __bf16* WpH  = (__bf16*)alloc((size_t)32 * 16 * 512 * 2);  // W_proj    [512x512],  KT=16, NT=32
  __bf16* WpL  = (__bf16*)alloc((size_t)32 * 16 * 512 * 2);
  __bf16* Wc1H = (__bf16*)alloc((size_t)32 * 28 * 512 * 2);  // [W_ih;W_hh] [896x512], KT=28, NT=32
  __bf16* Wc1L = (__bf16*)alloc((size_t)32 * 28 * 512 * 2);
  __bf16* Wc2H = (__bf16*)alloc((size_t)80 * 20 * 512 * 2);  // [W_l;W_r;W_t] [640x1280], KT=20, NT=80
  __bf16* Wc2L = (__bf16*)alloc((size_t)80 * 20 * 512 * 2);
  // per-step activation fragments
  __bf16* AtH = (__bf16*)alloc((size_t)16 * 28 * 512 * 2);
  __bf16* AtL = (__bf16*)alloc((size_t)16 * 28 * 512 * 2);
  __bf16* ArH = (__bf16*)alloc((size_t)16 * 20 * 512 * 2);
  __bf16* ArL = (__bf16*)alloc((size_t)16 * 20 * 512 * 2);

  // --- one-time weight pre-swizzle ---
  k_swizzle_w<<<64,  256, 0, stream>>>(W_proj,  512, 512,  16, 0,  WpH,  WpL);
  k_swizzle_w<<<96,  256, 0, stream>>>(W_ih,    768, 512,  28, 0,  Wc1H, Wc1L);
  k_swizzle_w<<<16,  256, 0, stream>>>(W_hh,    128, 512,  28, 24, Wc1H, Wc1L);
  k_swizzle_w<<<80,  256, 0, stream>>>(W_left,  256, 1280, 20, 0,  Wc2H, Wc2L);
  k_swizzle_w<<<80,  256, 0, stream>>>(W_right, 256, 1280, 20, 8,  Wc2H, Wc2L);
  k_swizzle_w<<<40,  256, 0, stream>>>(W_track, 128, 1280, 20, 16, Wc2H, Wc2L);

  // --- init + projection ---
  k_init<<<BATCH, TTRACK, 0, stream>>>(hbuf, cbuf, context_begin, state, loss);
  k_proj<<<2048, 256, 0, stream>>>(sentences, WpH, WpL, b_proj, buffers);

  // --- 255 sequential shift/reduce steps ---
  for (int t = 0; t < NSTEPS; ++t) {
    const int* actPtr = gold + (size_t)t * BATCH;
    k_swizA_tracker<<<56, 256, 0, stream>>>(buffers, sentence_end, stack_base, composed,
                                            hbuf, state, AtH, AtL);
    k_gemm_tdm<28><<<64, 256, 0, stream>>>(AtH, AtL, Wc1H, Wc1L, 16, 32,
                                           b_ih, b_hh, gates, 512, actPtr, -1);
    k_update<<<BATCH, TTRACK, 0, stream>>>(gates, hbuf, cbuf, W_trans, b_trans, actPtr, loss);
    k_swizA_reduce<<<40, 256, 0, stream>>>(buffers, stack_base, composed, hbuf, state,
                                           actPtr, ArH, ArL);
    k_gemm_tdm<20><<<160, 256, 0, stream>>>(ArH, ArL, Wc2H, Wc2L, 16, 80,
                                            b_left, nullptr, lstm_in, 1280, actPtr, ACT_REDUCE);
    k_compose<<<BATCH, DMODEL, 0, stream>>>(lstm_in, buffers, stack_base, composed, state, actPtr);
    k_book<<<1, 1, 0, stream>>>(state, actPtr);
  }

  // --- output: final_h [256,256] + loss/(255*256) ---
  k_final<<<BATCH, DMODEL, 0, stream>>>(state, buffers, stack_base, composed, loss, (float*)d_out);
}
